// MultiHeadAttention_42855183679890
// MI455X (gfx1250) — compile-verified, matrix operand order checked by reading, not run on hardware
//
#include <hip/hip_runtime.h>
#include <hip/hip_bf16.h>
#include <math.h>

// ---------------- problem constants (from reference) ----------------
#define Nb    2
#define Tdim  2048
#define Ddim  1024
#define Hn    16
#define HDd   64
#define WL    127
#define WR    128
#define PAD_L 128
#define T_PAD 2336   // 128 left pad + 2048 + 160 right pad (covers sb up to t0+159)

typedef __attribute__((ext_vector_type(16))) __bf16 v16bf;
typedef __attribute__((ext_vector_type(8)))  __bf16 v8bfv;
typedef __attribute__((ext_vector_type(8)))  float  v8f;

#if __has_builtin(__builtin_amdgcn_global_load_async_to_lds_b128)
#define HAVE_ASYNC_LDS 1
#else
#define HAVE_ASYNC_LDS 0
#endif

__device__ __forceinline__ v8f wmma_bf16(v16bf a, v16bf b, v8f c) {
  return __builtin_amdgcn_wmma_f32_16x16x32_bf16(false, a, false, b, (short)0, c, false, false);
}

// Load a 16x32 bf16 A/B-operand fragment from a row-major matrix.
// CDNA5 16-bit operand layout: lane (l&15) = row; koff=(l>>4)*8 selects K chunks
// [koff,koff+8) in elements 0..7 and [koff+16,koff+24) in elements 8..15.
__device__ __forceinline__ v16bf load_frag(const __bf16* __restrict__ p, int ld,
                                           int lane, int kbase) {
  const int r    = lane & 15;
  const int koff = (lane >> 4) << 3;
  const __bf16* base = p + (size_t)r * ld + kbase + koff;
  union { v16bf v; v8bfv h[2]; } u;
  u.h[0] = *(const v8bfv*)(base);
  u.h[1] = *(const v8bfv*)(base + 16);
  return u.v;
}

#if HAVE_ASYNC_LDS
// Pointee type must match the builtin prototype exactly: int vector_size(16).
typedef int v4i_vs __attribute__((vector_size(16)));
typedef v4i_vs __attribute__((address_space(1)))* gv4p;
typedef v4i_vs __attribute__((address_space(3)))* lv4p;

// Stage a 32-row x 64-col bf16 tile (row stride 128B in BOTH global and LDS):
// lane = row, 8 async b128 ops with immediate 16B chunk offsets.
__device__ __forceinline__ void stage_k_async(const __bf16* kp, __bf16* lbuf, int lane) {
  gv4p g = (gv4p)(uintptr_t)(kp + lane * 64);
  lv4p l = (lv4p)(unsigned)(uintptr_t)(lbuf + lane * 64);
  __builtin_amdgcn_global_load_async_to_lds_b128(g, l,   0, 0);
  __builtin_amdgcn_global_load_async_to_lds_b128(g, l,  16, 0);
  __builtin_amdgcn_global_load_async_to_lds_b128(g, l,  32, 0);
  __builtin_amdgcn_global_load_async_to_lds_b128(g, l,  48, 0);
  __builtin_amdgcn_global_load_async_to_lds_b128(g, l,  64, 0);
  __builtin_amdgcn_global_load_async_to_lds_b128(g, l,  80, 0);
  __builtin_amdgcn_global_load_async_to_lds_b128(g, l,  96, 0);
  __builtin_amdgcn_global_load_async_to_lds_b128(g, l, 112, 0);
}

__device__ __forceinline__ void wait_async8() {
#if __has_builtin(__builtin_amdgcn_s_wait_asynccnt)
  __builtin_amdgcn_s_wait_asynccnt(8);
#else
  asm volatile("s_wait_asynccnt 0x8" ::: "memory");
#endif
}
__device__ __forceinline__ void wait_async0() {
#if __has_builtin(__builtin_amdgcn_s_wait_asynccnt)
  __builtin_amdgcn_s_wait_asynccnt(0);
#else
  asm volatile("s_wait_asynccnt 0x0" ::: "memory");
#endif
}
#endif // HAVE_ASYNC_LDS

// ---------------- kernel 0: f32 -> bf16 pack ----------------
__global__ void k_tobf16(const float* __restrict__ s, __bf16* __restrict__ d, int n) {
  int i = blockIdx.x * blockDim.x + threadIdx.x;
  if (i < n) d[i] = (__bf16)s[i];
}

// ---------------- bf16 WMMA GEMM: C[M x Nc] = A[M x K] * B[Nc x K]^T (+bias) ----------------
// One wave computes a 32(M) x 64(N) tile: 8 accumulators; per k-step 2 A-frags,
// 4 B-frags, 8 WMMAs -> halves L2 traffic vs a 16x64 tile.
__global__ void __launch_bounds__(256)
k_gemm_bf16(const __bf16* __restrict__ A, const __bf16* __restrict__ B,
            float* __restrict__ C, const float* __restrict__ bias,
            int M, int Nc, int K) {
  const int lane = threadIdx.x & 31;
  const int wid  = blockIdx.x * (blockDim.x >> 5) + (threadIdx.x >> 5);
  const int ctiles = Nc >> 6;
  const int mt = wid / ctiles;
  const int nt = wid % ctiles;
  if (mt * 32 >= M) return;

  const __bf16* a0 = A + (size_t)(mt * 32 +  0) * K;
  const __bf16* a1 = A + (size_t)(mt * 32 + 16) * K;
  const __bf16* b0 = B + (size_t)(nt * 64 +  0) * K;
  const __bf16* b1 = B + (size_t)(nt * 64 + 16) * K;
  const __bf16* b2 = B + (size_t)(nt * 64 + 32) * K;
  const __bf16* b3 = B + (size_t)(nt * 64 + 48) * K;

  v8f acc[2][4] = {};
  for (int kb = 0; kb < K; kb += 32) {
    v16bf af0 = load_frag(a0, K, lane, kb);
    v16bf af1 = load_frag(a1, K, lane, kb);
    v16bf f0  = load_frag(b0, K, lane, kb);
    v16bf f1  = load_frag(b1, K, lane, kb);
    v16bf f2  = load_frag(b2, K, lane, kb);
    v16bf f3  = load_frag(b3, K, lane, kb);
    acc[0][0] = wmma_bf16(af0, f0, acc[0][0]);
    acc[0][1] = wmma_bf16(af0, f1, acc[0][1]);
    acc[0][2] = wmma_bf16(af0, f2, acc[0][2]);
    acc[0][3] = wmma_bf16(af0, f3, acc[0][3]);
    acc[1][0] = wmma_bf16(af1, f0, acc[1][0]);
    acc[1][1] = wmma_bf16(af1, f1, acc[1][1]);
    acc[1][2] = wmma_bf16(af1, f2, acc[1][2]);
    acc[1][3] = wmma_bf16(af1, f3, acc[1][3]);
  }

  const int colb = nt * 64;
  const int r0   = (lane >> 4) << 3;
  const int cc   = lane & 15;
  float bb0 = bias ? bias[colb +  0 + cc] : 0.f;
  float bb1 = bias ? bias[colb + 16 + cc] : 0.f;
  float bb2 = bias ? bias[colb + 32 + cc] : 0.f;
  float bb3 = bias ? bias[colb + 48 + cc] : 0.f;
#pragma unroll
  for (int g = 0; g < 2; ++g) {
#pragma unroll
    for (int i = 0; i < 8; ++i) {
      int m = mt * 32 + g * 16 + r0 + i;
      float* crow = C + (size_t)m * Nc + colb + cc;
      crow[ 0] = acc[g][0][i] + bb0;
      crow[16] = acc[g][1][i] + bb1;
      crow[32] = acc[g][2][i] + bb2;
      crow[48] = acc[g][3][i] + bb3;
    }
  }
}

// ---------------- RoPE + layout pack ----------------
// qkv f32 [n][t][3][h][hd] -> qb/kb bf16 [n][h][t_pad][64] (RoPE'd, q pre-scaled 1/8),
// vtb bf16 [n][h][64][t_pad] (transposed). Pad regions zero-filled.
__global__ void k_rope_pack(const float* __restrict__ qkv,
                            __bf16* __restrict__ qb, __bf16* __restrict__ kb,
                            __bf16* __restrict__ vtb) {
  size_t id = (size_t)blockIdx.x * blockDim.x + threadIdx.x;
  int hd = (int)(id & 63);
  size_t tmp = id >> 6;
  int tp = (int)(tmp % T_PAD);
  size_t rest = tmp / T_PAD;
  int h = (int)(rest & 15);
  int n = (int)(rest >> 4);
  if (n >= Nb) return;

  size_t qidx = ((size_t)(n * Hn + h) * T_PAD + tp) * HDd + hd;
  size_t vidx = ((size_t)(n * Hn + h) * HDd + hd) * T_PAD + tp;
  int t = tp - PAD_L;
  if (t < 0 || t >= Tdim) {
    qb[qidx] = (__bf16)0.f; kb[qidx] = (__bf16)0.f; vtb[vidx] = (__bf16)0.f;
    return;
  }
  const float* base = qkv + (size_t)(n * Tdim + t) * 3 * Ddim;
  int off = h * HDd;
  float qv = base[off + hd];
  float kv = base[Ddim + off + hd];
  float vv = base[2 * Ddim + off + hd];

  int i = hd & 31;
  // inv_freq = 10000^(-i/32) = exp(-i * ln(10000)/32)
  float inv = __expf(-0.28782313662425572f * (float)i);
  float ang = (float)t * inv;
  float c = cosf(ang), s = sinf(ang);
  float qo, ko;
  if (hd < 32) {
    float q2 = base[off + hd + 32], k2 = base[Ddim + off + hd + 32];
    qo = qv * c - q2 * s;
    ko = kv * c - k2 * s;
  } else {
    float q1 = base[off + hd - 32], k1 = base[Ddim + off + hd - 32];
    qo = qv * c + q1 * s;
    ko = kv * c + k1 * s;
  }
  qb[qidx]  = (__bf16)(qo * 0.125f);   // fold 1/sqrt(64)
  kb[qidx]  = (__bf16)ko;
  vtb[vidx] = (__bf16)vv;
}

// ---------------- banded flash attention ----------------
// One wave per (n, h, 16-query tile). 9 steps of 32 keys cover [t0-128, t0+160).
// K tiles are double-buffer staged into LDS with async-to-LDS ops when available.
__global__ void __launch_bounds__(256)
k_attn(const __bf16* __restrict__ qb, const __bf16* __restrict__ kb,
       const __bf16* __restrict__ vtb, __bf16* __restrict__ ctx) {
  __shared__ __bf16 lds_p[8][16 * 32];
#if HAVE_ASYNC_LDS
  __shared__ __bf16 lds_k[8][2][32 * 64];
#endif

  const int lane   = threadIdx.x & 31;
  const int wlocal = threadIdx.x >> 5;
  const int wid    = blockIdx.x * 8 + wlocal;
  const int ttile  = wid & 127;
  const int h      = (wid >> 7) & 15;
  const int n      = wid >> 11;
  const int t0     = ttile * 16;

  const __bf16* qp = qb + ((size_t)(n * Hn + h) * T_PAD + PAD_L + t0) * HDd;
  v16bf qa0 = load_frag(qp, HDd, lane, 0);
  v16bf qa1 = load_frag(qp, HDd, lane, 32);

  float mr[8], lr[8];
#pragma unroll
  for (int i = 0; i < 8; ++i) { mr[i] = -1e30f; lr[i] = 0.f; }
  v8f o0 = {}, o1 = {}, o2 = {}, o3 = {};

  const int r0 = (lane >> 4) << 3;
  const int cc = lane & 15;
  const __bf16* kbp = kb  + (size_t)(n * Hn + h) * T_PAD * HDd;
  const __bf16* vbp = vtb + (size_t)(n * Hn + h) * HDd * T_PAD;
  __bf16* prow = &lds_p[wlocal][0];

#if HAVE_ASYNC_LDS
  // prefetch step 0's 32x64 K tile
  stage_k_async(kbp + (size_t)(PAD_L + t0 - 128) * HDd, &lds_k[wlocal][0][0], lane);
#endif

  for (int step = 0; step < 9; ++step) {
    int sb = t0 - 128 + step * 32;
    v16bf kA0, kA1, kB0, kB1;
#if HAVE_ASYNC_LDS
    if (step < 8)
      stage_k_async(kbp + (size_t)(PAD_L + sb + 32) * HDd,
                    &lds_k[wlocal][(step + 1) & 1][0], lane);
    if (step < 8) wait_async8(); else wait_async0();
    __builtin_amdgcn_wave_barrier();
    const __bf16* lk = &lds_k[wlocal][step & 1][0];
    kA0 = load_frag(lk,           64, lane, 0);
    kA1 = load_frag(lk,           64, lane, 32);
    kB0 = load_frag(lk + 16 * 64, 64, lane, 0);
    kB1 = load_frag(lk + 16 * 64, 64, lane, 32);
#else
    const __bf16* kp = kbp + (size_t)(PAD_L + sb) * HDd;
    kA0 = load_frag(kp,            HDd, lane, 0);
    kA1 = load_frag(kp,            HDd, lane, 32);
    kB0 = load_frag(kp + 16 * HDd, HDd, lane, 0);
    kB1 = load_frag(kp + 16 * HDd, HDd, lane, 32);
#endif

    v8f s0 = {}, s1 = {};
    s0 = wmma_bf16(qa0, kA0, s0);
    s0 = wmma_bf16(qa1, kA1, s0);
    s1 = wmma_bf16(qa0, kB0, s1);
    s1 = wmma_bf16(qa1, kB1, s1);

#pragma unroll
    for (int i = 0; i < 8; ++i) {
      int tq = t0 + r0 + i;
      int sA = sb + cc, sB = sb + 16 + cc;
      bool okA = (sA >= tq - WL) && (sA <= tq + WR) && (sA >= 0) && (sA < Tdim);
      bool okB = (sB >= tq - WL) && (sB <= tq + WR) && (sB >= 0) && (sB < Tdim);
      float vA = okA ? s0[i] : -1e30f;
      float vB = okB ? s1[i] : -1e30f;
      float tm = fmaxf(vA, vB);
      tm = fmaxf(tm, __shfl_xor(tm, 1, 32));
      tm = fmaxf(tm, __shfl_xor(tm, 2, 32));
      tm = fmaxf(tm, __shfl_xor(tm, 4, 32));
      tm = fmaxf(tm, __shfl_xor(tm, 8, 32));
      float nm    = fmaxf(mr[i], tm);
      float alpha = __expf(mr[i] - nm);
      float pA = okA ? __expf(vA - nm) : 0.f;
      float pB = okB ? __expf(vB - nm) : 0.f;
      float rs = pA + pB;
      rs += __shfl_xor(rs, 1, 32);
      rs += __shfl_xor(rs, 2, 32);
      rs += __shfl_xor(rs, 4, 32);
      rs += __shfl_xor(rs, 8, 32);
      lr[i] = lr[i] * alpha + rs;
      mr[i] = nm;
      o0[i] *= alpha; o1[i] *= alpha; o2[i] *= alpha; o3[i] *= alpha;
      prow[(r0 + i) * 32 + cc]      = (__bf16)pA;
      prow[(r0 + i) * 32 + 16 + cc] = (__bf16)pB;
    }
    __builtin_amdgcn_wave_barrier();   // keep LDS P writes ordered before reads (single wave)

    v16bf pa = load_frag(prow, 32, lane, 0);
    const __bf16* vp = vbp + PAD_L + sb;
    v16bf vf0 = load_frag(vp + (size_t) 0 * T_PAD, T_PAD, lane, 0);
    v16bf vf1 = load_frag(vp + (size_t)16 * T_PAD, T_PAD, lane, 0);
    v16bf vf2 = load_frag(vp + (size_t)32 * T_PAD, T_PAD, lane, 0);
    v16bf vf3 = load_frag(vp + (size_t)48 * T_PAD, T_PAD, lane, 0);
    o0 = wmma_bf16(pa, vf0, o0);
    o1 = wmma_bf16(pa, vf1, o1);
    o2 = wmma_bf16(pa, vf2, o2);
    o3 = wmma_bf16(pa, vf3, o3);
  }

#pragma unroll
  for (int i = 0; i < 8; ++i) {
    float inv = 1.f / lr[i];
    int t = t0 + r0 + i;
    __bf16* crow = ctx + (size_t)(n * Tdim + t) * Ddim + h * HDd;
    crow[ 0 + cc] = (__bf16)(o0[i] * inv);
    crow[16 + cc] = (__bf16)(o1[i] * inv);
    crow[32 + cc] = (__bf16)(o2[i] * inv);
    crow[48 + cc] = (__bf16)(o3[i] * inv);
  }
}

// ---------------- host-side launch ----------------
extern "C" void kernel_launch(void* const* d_in, const int* in_sizes, int n_in,
                              void* d_out, int out_size, void* d_ws, size_t ws_size,
                              hipStream_t stream) {
  const float* x     = (const float*)d_in[0];   // [N,T,D]
  const float* wqkv  = (const float*)d_in[1];   // [3D,D]
  const float* wout  = (const float*)d_in[2];   // [D,D]
  const float* bout  = (const float*)d_in[3];   // [D]
  float* out = (float*)d_out;                   // [N,T,D] f32

  const size_t M   = (size_t)Nb * Tdim;         // 4096
  const size_t nx  = M * Ddim;                  // 4M
  const size_t nwq = (size_t)3 * Ddim * Ddim;   // 3M
  const size_t nwo = (size_t)Ddim * Ddim;       // 1M
  const size_t npk = (size_t)Nb * Hn * T_PAD * HDd;

  char* ws = (char*)d_ws;
  size_t off = 0;
  auto alloc = [&](size_t bytes) { char* p = ws + off; off += (bytes + 255) & ~(size_t)255; return p; };
  __bf16* xb    = (__bf16*)alloc(nx  * 2);
  __bf16* wqkvb = (__bf16*)alloc(nwq * 2);
  __bf16* woutb = (__bf16*)alloc(nwo * 2);
  float*  qkvf  = (float*) alloc(M * 3 * Ddim * 4);
  __bf16* qb    = (__bf16*)alloc(npk * 2);
  __bf16* kbuf  = (__bf16*)alloc(npk * 2);
  __bf16* vtb   = (__bf16*)alloc(npk * 2);
  __bf16* ctx   = (__bf16*)alloc(nx * 2);
  (void)ws_size; (void)in_sizes; (void)n_in; (void)out_size;

  // 1. pack to bf16
  k_tobf16<<<(int)((nx  + 255) / 256), 256, 0, stream>>>(x,    xb,    (int)nx);
  k_tobf16<<<(int)((nwq + 255) / 256), 256, 0, stream>>>(wqkv, wqkvb, (int)nwq);
  k_tobf16<<<(int)((nwo + 255) / 256), 256, 0, stream>>>(wout, woutb, (int)nwo);

  // 2. qkv = x @ wqkv^T   (M=4096, Nc=3072, K=1024): 128*48 = 6144 waves -> 768 blocks
  k_gemm_bf16<<<768, 256, 0, stream>>>(xb, wqkvb, qkvf, nullptr,
                                       (int)M, 3 * Ddim, Ddim);

  // 3. RoPE + pack q/k/v layouts (covers zero padding)
  {
    size_t tot = (size_t)Nb * Hn * T_PAD * HDd;
    k_rope_pack<<<(int)((tot + 255) / 256), 256, 0, stream>>>(qkvf, qb, kbuf, vtb);
  }

  // 4. banded attention: 2*16*128 = 4096 waves -> 512 blocks
  k_attn<<<512, 256, 0, stream>>>(qb, kbuf, vtb, ctx);

  // 5. out = ctx @ wout^T + bout  (M=4096, Nc=1024, K=1024): 128*16 = 2048 waves -> 256 blocks
  k_gemm_bf16<<<256, 256, 0, stream>>>(ctx, woutb, out, bout,
                                       (int)M, Ddim, Ddim);
}
